// DoubleAttention_18915035972049
// MI455X (gfx1250) — matrix operand order repaired and checked by reading, not compiled, not run on hardware
//
#include <hip/hip_runtime.h>
#include <hip/hip_bf16.h>
#include <cfloat>
#include <cmath>

// Problem constants (match reference)
#define BN    2048              // N query tokens
#define BNC   2048              // NC context tokens
#define BDIM  1024              // model dim
#define BH    16                // heads
#define BD    64                // head dim
#define BM    16                // memory kv slots
#define BNK   (BNC + BM)        // 2064 total keys
#define NKP   2080              // padded key stride (= 65*32), pad cols kept at 0
#define BTOPK 64

typedef __bf16 bf16_t;
typedef __attribute__((ext_vector_type(16))) __bf16 v16bf;
typedef __attribute__((ext_vector_type(8)))  __bf16 v8bf;
typedef __attribute__((ext_vector_type(8)))  float  v8f;
typedef __attribute__((ext_vector_type(4)))  float  v4f;
typedef __attribute__((ext_vector_type(4)))  unsigned int v4u;
typedef __attribute__((ext_vector_type(8)))  int v8i;
typedef __attribute__((ext_vector_type(4)))  int v4i;

// ---------------------------------------------------------------------------
// WMMA fragment loaders (CDNA5 ISA 7.12.2 layouts, wave32) — unguarded,
// vectorized: each lane issues two 16B (bf16) / four 16B (f32) loads.
// ---------------------------------------------------------------------------

// A 16x32 bf16 from row-major [row][k] (stride lda). Lane r=lane&15 -> row M=r;
// half=lane>>4: elements 0..7 = K(half*8+0..7), 8..15 = K(16+half*8+0..7).
__device__ inline v16bf load_a_bf(const bf16_t* A, int lda, int row0, int k0) {
  int lane = threadIdx.x & 31;
  int half = lane >> 4;
  int r    = lane & 15;
  const bf16_t* p = A + (size_t)(row0 + r) * lda + k0 + half * 8;
  v8bf lo = *(const v8bf*)(p);
  v8bf hi = *(const v8bf*)(p + 16);
  v16bf out;
#pragma unroll
  for (int e = 0; e < 8; ++e) { out[e] = lo[e]; out[e + 8] = hi[e]; }
  return out;
}

// Same A layout, f32 source converted in-register (for attn@V).
__device__ inline v16bf load_a_f32(const float* A, int lda, int row0, int k0) {
  int lane = threadIdx.x & 31;
  int half = lane >> 4;
  int r    = lane & 15;
  const float* p = A + (size_t)(row0 + r) * lda + k0 + half * 8;
  v4f a0 = *(const v4f*)(p);
  v4f a1 = *(const v4f*)(p + 4);
  v4f b0 = *(const v4f*)(p + 16);
  v4f b1 = *(const v4f*)(p + 20);
  v16bf out;
#pragma unroll
  for (int e = 0; e < 4; ++e) {
    out[e]      = (bf16_t)a0[e];
    out[e + 4]  = (bf16_t)a1[e];
    out[e + 8]  = (bf16_t)b0[e];
    out[e + 12] = (bf16_t)b1[e];
  }
  return out;
}

// B 32x16 where memory holds B^T row-major: [n][k] (stride ldb).
// Lane column N=lane&15; half=lane>>4: elements 0..7 = K(half*8+0..7), 8..15 = K(16+half*8+0..7).
__device__ inline v16bf load_bt_bf(const bf16_t* Bt, int ldb, int k0, int col0) {
  int lane = threadIdx.x & 31;
  int n  = lane & 15;
  int kh = (lane >> 4) * 8;
  const bf16_t* p = Bt + (size_t)(col0 + n) * ldb + k0 + kh;
  v8bf lo = *(const v8bf*)(p);
  v8bf hi = *(const v8bf*)(p + 16);
  v16bf out;
#pragma unroll
  for (int e = 0; e < 8; ++e) { out[e] = lo[e]; out[e + 8] = hi[e]; }
  return out;
}

// B fragment out of an LDS-staged [n][32] tile (row stride 32 bf16 = 64B).
__device__ inline v16bf load_b_lds(const bf16_t* tile, int col0) {
  int lane = threadIdx.x & 31;
  int n  = lane & 15;
  int kh = (lane >> 4) * 8;
  const bf16_t* p = tile + (size_t)(col0 + n) * 32 + kh;
  v8bf lo = *(const v8bf*)(p);
  v8bf hi = *(const v8bf*)(p + 16);
  v16bf out;
#pragma unroll
  for (int e = 0; e < 8; ++e) { out[e] = lo[e]; out[e + 8] = hi[e]; }
  return out;
}

// ---------------------------------------------------------------------------
// Tensor Data Mover: async 2D tile load Bt[row0..row0+63][k0..k0+31] -> LDS.
// D# bitfields per CDNA5 ISA ch.8 (group0: count/lds/global/type; group1:
// data_size=2B, tile 32x64, tensor dims & dim0 stride). 2D -> groups 2/3
// unused (zero). This toolchain exposes the 6-arg builtin form.
// ---------------------------------------------------------------------------
__device__ inline void tdm_load_b(unsigned lds_off, const bf16_t* gaddr, int Kk, int Nn) {
  unsigned long long ga = (unsigned long long)(uintptr_t)gaddr;
  v4u g0;
  g0[0] = 1u;                                        // count=1, user, no gather
  g0[1] = lds_off;                                   // lds_addr (bytes)
  g0[2] = (unsigned)(ga & 0xFFFFFFFFull);            // global_addr[31:0]
  g0[3] = (unsigned)((ga >> 32) & 0x01FFFFFFull)     // global_addr[56:32]
        | (2u << 30);                                // type = 2 ("image")
  unsigned K = (unsigned)Kk, Nu = (unsigned)Nn;
  v8i g1;
  g1[0] = (int)(1u << 16);                           // data_size=1 (2 bytes), mask=0
  g1[1] = (int)((K & 0xFFFFu) << 16);                // tensor_dim0[15:0]
  g1[2] = (int)((K >> 16) | ((Nu & 0xFFFFu) << 16)); // tensor_dim0[31:16] | tensor_dim1[15:0]
  g1[3] = (int)((Nu >> 16) | (32u << 16));           // tensor_dim1[31:16] | tile_dim0=32
  g1[4] = (int)64;                                   // tile_dim1=64, tile_dim2=0
  g1[5] = (int)K;                                    // tensor_dim0_stride[31:0]
  g1[6] = 0;                                         // stride hi / dim1_stride lo
  g1[7] = 0;
  v4i gz4 = {};
  v8i gz8 = {};
  __builtin_amdgcn_tensor_load_to_lds(g0, g1, gz4, gz4, gz8, 0);
}

// ---------------------------------------------------------------------------
// Elementwise conversion kernels
// ---------------------------------------------------------------------------

__global__ void k_f32_to_bf16(const float* __restrict__ src, bf16_t* __restrict__ dst, int n) {
  int i = blockIdx.x * 256 + threadIdx.x;
  if (i < n) dst[i] = (bf16_t)src[i];
}

// src [R][C] f32 row-major -> dst [C][R] bf16 (transposed, for B^T GEMM feeds)
__global__ void k_cvt_t(const float* __restrict__ src, bf16_t* __restrict__ dst, int R, int C) {
  int idx = blockIdx.x * 256 + threadIdx.x;
  if (idx >= R * C) return;
  int r = idx / C, c = idx % C;
  dst[(size_t)c * R + r] = (bf16_t)src[idx];
}

// ---------------------------------------------------------------------------
// GEMM: C[M][N] f32 = A[M][K]bf16 * B[K][N]bf16, B stored transposed ([N][K]).
// One wave computes a 16x64 strip. B tiles (64x32) are DMA'd into LDS by the
// Tensor Data Mover, double-buffered, pipelined with s_wait_tensorcnt.
// ---------------------------------------------------------------------------
__global__ void __launch_bounds__(32) k_gemm(const bf16_t* __restrict__ A,
                                             const bf16_t* __restrict__ Bt,
                                             float* __restrict__ C,
                                             int Mm, int Nn, int Kk) {
  (void)Mm;
  __shared__ bf16_t ldsB[2][64 * 32];     // two 4KB B tiles
  int ng = Nn >> 6;                       // groups of 64 columns
  int tm = blockIdx.x / ng;
  int tg = blockIdx.x % ng;
  const bf16_t* bbase = Bt + (size_t)(tg * 64) * Kk;   // tile anchor: row tg*64, k=0
  unsigned lds0 = (unsigned)(uintptr_t)&ldsB[0][0];
  unsigned lds1 = (unsigned)(uintptr_t)&ldsB[1][0];
  v8f acc[4] = {};
  int nsteps = Kk >> 5;
  tdm_load_b(lds0, bbase, Kk, Nn);                     // prime buffer 0
  for (int s = 0; s < nsteps; ++s) {
    int k0 = s << 5;
    if (s + 1 < nsteps) {
      tdm_load_b((s & 1) ? lds0 : lds1, bbase + (k0 + 32), Kk, Nn);
      __builtin_amdgcn_s_wait_tensorcnt(1);            // current tile resident
    } else {
      __builtin_amdgcn_s_wait_tensorcnt(0);
    }
    v16bf a = load_a_bf(A, Kk, tm * 16, k0);
    const bf16_t* tile = (s & 1) ? &ldsB[1][0] : &ldsB[0][0];
#pragma unroll
    for (int t = 0; t < 4; ++t) {
      v16bf b = load_b_lds(tile, t * 16);
      acc[t] = __builtin_amdgcn_wmma_f32_16x16x32_bf16(false, a, false, b, (short)0, acc[t], false, false);
    }
  }
  int lane = threadIdx.x & 31;
#pragma unroll
  for (int t = 0; t < 4; ++t) {
    int col = tg * 64 + t * 16 + (lane & 15);
#pragma unroll
    for (int r = 0; r < 8; ++r) {
      int row = tm * 16 + r + 8 * (lane >> 4);
      C[(size_t)row * Nn + col] = acc[t][r];
    }
  }
}

// ---------------------------------------------------------------------------
// Normalization / assembly kernels
// ---------------------------------------------------------------------------

// l2-normalize each (i,h) 64-vector of Q, convert to bf16 (same [i][h*64+d] layout).
__global__ void __launch_bounds__(32) k_norm_q(const float* __restrict__ qf, bf16_t* __restrict__ qn) {
  int i = blockIdx.x / BH;
  int h = blockIdx.x % BH;
  int lane = threadIdx.x;
  const float* p = qf + (size_t)i * BDIM + h * BD;
  float a = p[lane], b = p[lane + 32];
  float ss = a * a + b * b;
#pragma unroll
  for (int m = 16; m > 0; m >>= 1) ss += __shfl_xor(ss, m, 32);
  float inv = 1.0f / fmaxf(sqrtf(ss), 1e-12f);
  bf16_t* q = qn + (size_t)i * BDIM + h * BD;
  q[lane]      = (bf16_t)(a * inv);
  q[lane + 32] = (bf16_t)(b * inv);
}

// K assembly: mem_k (j<M) + projected context K (j>=M), l2-normalized, bf16,
// layout [h][j][d] (rows contiguous in d -> B^T for the QK^T WMMA).
__global__ void __launch_bounds__(32) k_build_k(const float* __restrict__ kcf,
                                                const float* __restrict__ mem_k,
                                                bf16_t* __restrict__ kn) {
  int h = blockIdx.x / BNK;
  int j = blockIdx.x % BNK;
  int lane = threadIdx.x;
  float a, b;
  if (j < BM) {
    const float* p = mem_k + ((size_t)h * BM + j) * BD;
    a = p[lane]; b = p[lane + 32];
  } else {
    const float* p = kcf + (size_t)(j - BM) * BDIM + h * BD;
    a = p[lane]; b = p[lane + 32];
  }
  float ss = a * a + b * b;
#pragma unroll
  for (int m = 16; m > 0; m >>= 1) ss += __shfl_xor(ss, m, 32);
  float inv = 1.0f / fmaxf(sqrtf(ss), 1e-12f);
  bf16_t* o = kn + ((size_t)h * BNK + j) * BD;
  o[lane]      = (bf16_t)(a * inv);
  o[lane + 32] = (bf16_t)(b * inv);
}

// V assembly, stored TRANSPOSED: vt[h][d][jp] with jp stride NKP; pad columns
// (j >= BNK) written as zero so the padded AV K-loop is exact.
__global__ void __launch_bounds__(32) k_build_v(const float* __restrict__ vcf,
                                                const float* __restrict__ mem_v,
                                                bf16_t* __restrict__ vt) {
  int h = blockIdx.x / NKP;
  int j = blockIdx.x % NKP;
  int lane = threadIdx.x;
  float a = 0.0f, b = 0.0f;
  if (j < BM) {
    const float* p = mem_v + ((size_t)h * BM + j) * BD;
    a = p[lane]; b = p[lane + 32];
  } else if (j < BNK) {
    const float* p = vcf + (size_t)(j - BM) * BDIM + h * BD;
    a = p[lane]; b = p[lane + 32];
  }
  bf16_t* base = vt + (size_t)h * BD * NKP;
  base[(size_t)lane * NKP + j]        = (bf16_t)a;
  base[(size_t)(lane + 32) * NKP + j] = (bf16_t)b;
}

// ---------------------------------------------------------------------------
// dots[h][i][jp] = scale[h] * qn(i)·kn(j), masked to -FLT_MAX (row stride NKP).
// ---------------------------------------------------------------------------
__global__ void __launch_bounds__(32) k_dots(const bf16_t* __restrict__ qn,
                                             const bf16_t* __restrict__ kn,
                                             const float* __restrict__ scale_p,
                                             const unsigned char* __restrict__ qmask,
                                             const unsigned char* __restrict__ kmask,
                                             float* __restrict__ dots) {
  int h  = blockIdx.z;
  int ti = blockIdx.y;
  int tj = blockIdx.x;
  const bf16_t* qh = qn + h * BD;                  // rows stride BDIM
  const bf16_t* kh = kn + (size_t)h * BNK * BD;    // [j][d] == B^T
  v8f acc = {};
#pragma unroll
  for (int k0 = 0; k0 < BD; k0 += 32) {
    v16bf a = load_a_bf(qh, BDIM, ti * 16, k0);
    v16bf b = load_bt_bf(kh, BD, k0, tj * 16);
    acc = __builtin_amdgcn_wmma_f32_16x16x32_bf16(false, a, false, b, (short)0, acc, false, false);
  }
  float scale = 1.0f / fmaxf(__expf(scale_p[h]), 0.01f);
  int lane = threadIdx.x & 31;
  int col  = tj * 16 + (lane & 15);
  bool kvalid = (col < BM) ? true : (kmask[col - BM] != 0);
#pragma unroll
  for (int r = 0; r < 8; ++r) {
    int row = ti * 16 + r + 8 * (lane >> 4);
    bool valid = kvalid && (qmask[row] != 0);
    float v = acc[r] * scale;
    dots[((size_t)h * BN + row) * NKP + col] = valid ? v : -FLT_MAX;
  }
}

// ---------------------------------------------------------------------------
// Talking-heads: in-place mix across the 16 head planes at each (i,j).
// ---------------------------------------------------------------------------
__global__ void __launch_bounds__(256) k_th_mix(float* __restrict__ dots,
                                                const float* __restrict__ th) {
  size_t idx = (size_t)blockIdx.x * 256 + threadIdx.x;
  if (idx >= (size_t)BN * BNK) return;
  int i = (int)(idx / BNK);
  int j = (int)(idx % BNK);
  const size_t plane = (size_t)BN * NKP;
  size_t base = (size_t)i * NKP + j;
  float v[BH];
#pragma unroll
  for (int hh = 0; hh < BH; ++hh) v[hh] = dots[(size_t)hh * plane + base];
#pragma unroll
  for (int g = 0; g < BH; ++g) {
    float s = 0.0f;
#pragma unroll
    for (int hh = 0; hh < BH; ++hh) s += th[g * BH + hh] * v[hh];
    dots[(size_t)g * plane + base] = s;
  }
}

// ---------------------------------------------------------------------------
// Per (g,i) row: radix-select the 64th-largest logit, drop below it, softmax
// in place; zero the pad columns. One 256-thread block per row.
// ---------------------------------------------------------------------------
__device__ inline unsigned fkey(float f) {
  unsigned u = __float_as_uint(f);
  return (u & 0x80000000u) ? ~u : (u | 0x80000000u);
}
__device__ inline float unkey(unsigned k) {
  unsigned u = (k & 0x80000000u) ? (k & 0x7FFFFFFFu) : ~k;
  return __uint_as_float(u);
}

__global__ void __launch_bounds__(256) k_topk_softmax(float* __restrict__ dots) {
  __shared__ float row[BNK];
  __shared__ float redf[256];
  __shared__ int   redi[256];
  __shared__ unsigned s_prefix;
  __shared__ int s_want;
  float* rp = dots + (size_t)blockIdx.x * NKP;
  int t = threadIdx.x;
  for (int j = t; j < BNK; j += 256) row[j] = rp[j];
  if (t == 0) { s_prefix = 0u; s_want = BTOPK; }
  __syncthreads();
  // MSB-first radix selection of the TOPK-th largest sortable key.
  for (int bit = 31; bit >= 0; --bit) {
    unsigned want_top = (s_prefix >> bit) | 1u;
    int cnt = 0;
    for (int j = t; j < BNK; j += 256)
      if ((fkey(row[j]) >> bit) == want_top) cnt++;
    redi[t] = cnt;
    __syncthreads();
    for (int s = 128; s > 0; s >>= 1) { if (t < s) redi[t] += redi[t + s]; __syncthreads(); }
    if (t == 0) {
      if (redi[0] >= s_want) s_prefix |= (1u << bit);
      else                   s_want  -= redi[0];
    }
    __syncthreads();
  }
  float kth = unkey(s_prefix);
  // row max (max always survives top-k)
  float mx = -FLT_MAX;
  for (int j = t; j < BNK; j += 256) mx = fmaxf(mx, row[j]);
  redf[t] = mx;
  __syncthreads();
  for (int s = 128; s > 0; s >>= 1) { if (t < s) redf[t] = fmaxf(redf[t], redf[t + s]); __syncthreads(); }
  mx = redf[0];
  __syncthreads();
  float sum = 0.0f;
  for (int j = t; j < BNK; j += 256) {
    float v = row[j];
    float e = (v >= kth) ? __expf(v - mx) : 0.0f;
    row[j] = e;
    sum += e;
  }
  redf[t] = sum;
  __syncthreads();
  for (int s = 128; s > 0; s >>= 1) { if (t < s) redf[t] += redf[t + s]; __syncthreads(); }
  float inv = 1.0f / redf[0];
  for (int j = t; j < BNK; j += 256) rp[j] = row[j] * inv;
  for (int j = BNK + t; j < NKP; j += 256) rp[j] = 0.0f;   // zero pad columns
}

// ---------------------------------------------------------------------------
// out(i, h*64+d) = head_scale[h] * sum_j attn[h][i][j] * v[h][j][d].
// One wave computes a full 16x64 (i,d) strip; fragments are software-pipelined
// (next K-step loads issued before current WMMAs) to overlap vmem with math.
// ---------------------------------------------------------------------------
__global__ void __launch_bounds__(32) k_av(const float* __restrict__ attn,
                                           const bf16_t* __restrict__ vt,
                                           const float* __restrict__ head_scale,
                                           float* __restrict__ oc) {
  int ti = blockIdx.x;
  int h  = blockIdx.y;
  const float*  ah = attn + (size_t)h * BN * NKP;   // [i][jp]
  const bf16_t* vh = vt   + (size_t)h * BD * NKP;   // [d][jp] == B^T
  v8f acc[4] = {};
  v16bf a_cur = load_a_f32(ah, NKP, ti * 16, 0);
  v16bf b_cur[4];
#pragma unroll
  for (int t = 0; t < 4; ++t) b_cur[t] = load_bt_bf(vh, NKP, 0, t * 16);
  for (int k0 = 0; k0 < NKP; k0 += 32) {            // 65 exact steps, pad==0
    v16bf a_nxt = {};
    v16bf b_nxt[4] = {};
    if (k0 + 32 < NKP) {
      a_nxt = load_a_f32(ah, NKP, ti * 16, k0 + 32);
#pragma unroll
      for (int t = 0; t < 4; ++t) b_nxt[t] = load_bt_bf(vh, NKP, k0 + 32, t * 16);
    }
#pragma unroll
    for (int t = 0; t < 4; ++t)
      acc[t] = __builtin_amdgcn_wmma_f32_16x16x32_bf16(false, a_cur, false, b_cur[t], (short)0, acc[t], false, false);
    a_cur = a_nxt;
#pragma unroll
    for (int t = 0; t < 4; ++t) b_cur[t] = b_nxt[t];
  }
  float hs = head_scale[h];
  int lane = threadIdx.x & 31;
#pragma unroll
  for (int t = 0; t < 4; ++t) {
    int col = t * 16 + (lane & 15);
#pragma unroll
    for (int r = 0; r < 8; ++r) {
      int row = ti * 16 + r + 8 * (lane >> 4);
      oc[(size_t)row * BDIM + h * BD + col] = acc[t][r] * hs;
    }
  }
}

// ---------------------------------------------------------------------------
// Host orchestration
// ---------------------------------------------------------------------------

struct WSPtrs {
  bf16_t *xb, *cb, *wt, *qn, *kn, *vt, *ocb;
  float  *qf, *kcf, *vcf, *dots, *oc, *y1;
};

static void run_layer(const float* x, const float* ctx,
                      const unsigned char* qmask, const unsigned char* kmask,
                      const float* Wq, const float* Wk, const float* Wv,
                      const float* th_pre, const float* th_post,
                      const float* mem_k, const float* mem_v,
                      const float* scale_p, const float* head_scale,
                      const float* Wo, float* y, const WSPtrs& w, hipStream_t s) {
  const int NDIM = BN * BDIM;      // 2,097,152
  const int WDIM = BDIM * BDIM;    // 1,048,576
  k_f32_to_bf16<<<NDIM / 256, 256, 0, s>>>(x,   w.xb, NDIM);
  k_f32_to_bf16<<<(BNC * BDIM) / 256, 256, 0, s>>>(ctx, w.cb, BNC * BDIM);
  // Q = x @ Wq ; l2norm
  k_cvt_t<<<WDIM / 256, 256, 0, s>>>(Wq, w.wt, BDIM, BDIM);
  k_gemm<<<(BN / 16) * (BDIM / 64), 32, 0, s>>>(w.xb, w.wt, w.qf, BN, BDIM, BDIM);
  k_norm_q<<<BN * BH, 32, 0, s>>>(w.qf, w.qn);
  // K = ctx @ Wk
  k_cvt_t<<<WDIM / 256, 256, 0, s>>>(Wk, w.wt, BDIM, BDIM);
  k_gemm<<<(BNC / 16) * (BDIM / 64), 32, 0, s>>>(w.cb, w.wt, w.kcf, BNC, BDIM, BDIM);
  // V = ctx @ Wv
  k_cvt_t<<<WDIM / 256, 256, 0, s>>>(Wv, w.wt, BDIM, BDIM);
  k_gemm<<<(BNC / 16) * (BDIM / 64), 32, 0, s>>>(w.cb, w.wt, w.vcf, BNC, BDIM, BDIM);
  // prepend memory kv; normalize K; V transposed + padded
  k_build_k<<<BH * BNK, 32, 0, s>>>(w.kcf, mem_k, w.kn);
  k_build_v<<<BH * NKP, 32, 0, s>>>(w.vcf, mem_v, w.vt);
  // logits + mask
  k_dots<<<dim3(BNK / 16, BN / 16, BH), 32, 0, s>>>(w.qn, w.kn, scale_p, qmask, kmask, w.dots);
  // talking heads (pre), top-k + softmax (+ pad zeroing), talking heads (post)
  {
    int thg = (int)(((size_t)BN * BNK + 255) / 256);
    k_th_mix<<<thg, 256, 0, s>>>(w.dots, th_pre);
    k_topk_softmax<<<BH * BN, 256, 0, s>>>(w.dots);
    k_th_mix<<<thg, 256, 0, s>>>(w.dots, th_post);
  }
  // attn @ V (+ head scale), reshape to [i][h*64+d]
  k_av<<<dim3(BN / 16, BH), 32, 0, s>>>(w.dots, w.vt, head_scale, w.oc);
  // out @ Wo
  k_f32_to_bf16<<<NDIM / 256, 256, 0, s>>>(w.oc, w.ocb, NDIM);
  k_cvt_t<<<WDIM / 256, 256, 0, s>>>(Wo, w.wt, BDIM, BDIM);
  k_gemm<<<(BN / 16) * (BDIM / 64), 32, 0, s>>>(w.ocb, w.wt, y, BN, BDIM, BDIM);
}

extern "C" void kernel_launch(void* const* d_in, const int* in_sizes, int n_in,
                              void* d_out, int out_size, void* d_ws, size_t ws_size,
                              hipStream_t stream) {
  (void)in_sizes; (void)n_in; (void)out_size; (void)ws_size;
  const float* x    = (const float*)d_in[0];
  const float* ctx  = (const float*)d_in[1];
  const unsigned char* qmask = (const unsigned char*)d_in[2];
  const unsigned char* kmask = (const unsigned char*)d_in[3];
  const float* Wq1 = (const float*)d_in[4];
  const float* Wk1 = (const float*)d_in[5];
  const float* Wv1 = (const float*)d_in[6];
  const float* Wq2 = (const float*)d_in[7];
  const float* Wk2 = (const float*)d_in[8];
  const float* Wv2 = (const float*)d_in[9];
  const float* thp1 = (const float*)d_in[10];
  const float* tho1 = (const float*)d_in[11];
  const float* thp2 = (const float*)d_in[12];
  const float* tho2 = (const float*)d_in[13];
  const float* mk1 = (const float*)d_in[14];
  const float* mv1 = (const float*)d_in[15];
  const float* mk2 = (const float*)d_in[16];
  const float* mv2 = (const float*)d_in[17];
  const float* sc1 = (const float*)d_in[18];
  const float* sc2 = (const float*)d_in[19];
  const float* hs1 = (const float*)d_in[20];
  const float* hs2 = (const float*)d_in[21];
  const float* Wo1 = (const float*)d_in[22];
  const float* Wo2 = (const float*)d_in[23];

  char* ws = (char*)d_ws;
  size_t off = 0;
  auto take = [&](size_t bytes) -> char* {
    char* p = ws + off;
    off += (bytes + 255) & ~(size_t)255;
    return p;
  };
  WSPtrs w;
  w.xb   = (bf16_t*)take((size_t)BN  * BDIM * 2);
  w.cb   = (bf16_t*)take((size_t)BNC * BDIM * 2);
  w.wt   = (bf16_t*)take((size_t)BDIM * BDIM * 2);
  w.qf   = (float*) take((size_t)BN  * BDIM * 4);
  w.qn   = (bf16_t*)take((size_t)BN  * BDIM * 2);
  w.kcf  = (float*) take((size_t)BNC * BDIM * 4);
  w.vcf  = (float*) take((size_t)BNC * BDIM * 4);
  w.kn   = (bf16_t*)take((size_t)BH * BNK * BD * 2);
  w.vt   = (bf16_t*)take((size_t)BH * BD * NKP * 2);
  w.dots = (float*) take((size_t)BH * BN * NKP * 4);   // ~272.6 MB
  w.oc   = (float*) take((size_t)BN * BDIM * 4);
  w.ocb  = (bf16_t*)take((size_t)BN * BDIM * 2);
  w.y1   = (float*) take((size_t)BN * BDIM * 4);

  // Layer 1: ctx slice 0 -> y1
  run_layer(x, ctx, qmask, kmask, Wq1, Wk1, Wv1, thp1, tho1, mk1, mv1,
            sc1, hs1, Wo1, w.y1, w, stream);
  // Layer 2: ctx slice 1, input = y1 -> d_out
  run_layer(w.y1, ctx + (size_t)BNC * BDIM, qmask, kmask, Wq2, Wk2, Wv2,
            thp2, tho2, mk2, mv2, sc2, hs2, Wo2, (float*)d_out, w, stream);
}
